// RGCN_66949950210089
// MI455X (gfx1250) — compile-verified, matrix operand order checked by reading, not run on hardware
//
#include <hip/hip_runtime.h>

#define Nn 50000
#define Ee 600000
#define Dd 128
#define Rr 8

typedef __attribute__((ext_vector_type(16))) __bf16 v16bf;
typedef __attribute__((ext_vector_type(8)))  __bf16 v8bf;
typedef __attribute__((ext_vector_type(8)))  float  v8f;

// ---------------- utility kernels ----------------

__global__ void zero_f32(float* __restrict__ p, int n) {
  int i = blockIdx.x * blockDim.x + threadIdx.x;
  if (i < n) p[i] = 0.0f;
}

__global__ void count_kernel(const int* __restrict__ et, const int* __restrict__ dst,
                             float* __restrict__ counts) {
  int e = blockIdx.x * blockDim.x + threadIdx.x;
  if (e < Ee) atomicAdd(&counts[et[e] * Nn + dst[e]], 1.0f);
}

__global__ void norm_kernel(const int* __restrict__ et, const int* __restrict__ dst,
                            const float* __restrict__ counts, float* __restrict__ nrm) {
  int e = blockIdx.x * blockDim.x + threadIdx.x;
  if (e < Ee) nrm[e] = 1.0f / fmaxf(counts[et[e] * Nn + dst[e]], 1.0f);
}

__global__ void cvt_bf16(const float* __restrict__ s, __bf16* __restrict__ d, int n) {
  int i = blockIdx.x * blockDim.x + threadIdx.x;
  if (i < n) d[i] = (__bf16)s[i];
}

// Repack a 128x128 f32 matrix (B[k][n], row-major) into WMMA B-fragment order:
// bf16 flat index = (((kc*8 + nt)*32) + lane)*16 + h, so each lane's v16bf
// fragment is one contiguous 32-byte chunk.
__global__ void repack_frag(const float* __restrict__ src, __bf16* __restrict__ dstf) {
  int idx = blockIdx.x * 256 + threadIdx.x;          // 0..16383
  const float* s = src + (size_t)blockIdx.y * 16384;
  __bf16* d = dstf + (size_t)blockIdx.y * 16384;
  int h    = idx & 15;
  int lane = (idx >> 4) & 31;
  int nt   = (idx >> 9) & 7;
  int kc   = idx >> 12;
  int kg   = (lane >> 4) * 8;
  int kk   = (h < 8) ? (kg + h) : (16 + kg + (h - 8));
  int k    = kc * 32 + kk;
  int n    = nt * 16 + (lane & 15);
  d[idx] = (__bf16)s[k * 128 + n];
}

// ---------------- WMMA GEMM ----------------
// C[M,128] = A[M,128] (bf16 row-major) x Bfrag (fragment-ordered bf16 128x128)
// B is staged once per block into LDS (32 KB), so the WMMA K-loop feeds from
// ds_load instead of long-latency global loads, and global B traffic drops 8x.
// MODE 0: out = acc
// MODE 1: out = acc + bias[col]
// MODE 2: out = res + relu(acc + agg + bias[col])
template <int MODE>
__global__ __launch_bounds__(256) void gemm_wmma(
    const __bf16* __restrict__ A, const __bf16* __restrict__ Bf,
    float* __restrict__ out, const float* __restrict__ agg,
    const float* __restrict__ bias, const float* __restrict__ res, int M) {
  __shared__ __align__(32) __bf16 Bs[16384];         // 32 KB fragment-ordered B

  // Cooperative stage: 2048 float4 chunks, 8 per thread.
  {
    const float4* gsrc = (const float4*)Bf;
    float4* ldst = (float4*)Bs;
#pragma unroll
    for (int i = 0; i < 8; ++i)
      ldst[threadIdx.x + 256 * i] = gsrc[threadIdx.x + 256 * i];
  }
  __syncthreads();                                    // all waves participate

  int wave = threadIdx.x >> 5;
  int lane = threadIdx.x & 31;
  int strip = blockIdx.x * 8 + wave;
  if (strip * 16 >= M) return;                        // tail waves exit AFTER barrier
  int m0 = strip * 16;

  // A fragment addressing (16-bit A 16x32 layout): lane half selects K-group.
  int kg = (lane >> 4) * 8;
  int arow = m0 + (lane & 15);
  const __bf16* Ap = A + (size_t)arow * 128 + kg;

  v8f acc[8] = {};

#pragma unroll
  for (int kc = 0; kc < 4; ++kc) {
    v8bf alo = *(const v8bf*)(Ap + kc * 32);
    v8bf ahi = *(const v8bf*)(Ap + kc * 32 + 16);
    v16bf a = __builtin_shufflevector(alo, ahi, 0, 1, 2, 3, 4, 5, 6, 7,
                                      8, 9, 10, 11, 12, 13, 14, 15);
#pragma unroll
    for (int nt = 0; nt < 8; ++nt) {
      v16bf b = *(const v16bf*)(Bs + ((kc * 8 + nt) * 32 + lane) * 16);
      acc[nt] = __builtin_amdgcn_wmma_f32_16x16x32_bf16(
          false, a, false, b, (short)0, acc[nt], false, false);
    }
  }

  // C layout: VGPR j -> row m0 + j (lanes 0-15) or m0 + 8 + j (lanes 16-31),
  // column = nt*16 + (lane & 15)
  int roff = (lane < 16) ? 0 : 8;
  int cl = lane & 15;
#pragma unroll
  for (int nt = 0; nt < 8; ++nt) {
    int col = nt * 16 + cl;
#pragma unroll
    for (int j = 0; j < 8; ++j) {
      size_t o = (size_t)(m0 + roff + j) * 128 + col;
      float v = acc[nt][j];
      if (MODE == 1) v += bias[col];
      if (MODE == 2) v = res[o] + fmaxf(v + agg[o] + bias[col], 0.0f);
      out[o] = v;
    }
  }
}

// ---------------- per-edge gather + scatter-add ----------------
// One wave per edge: 128 floats -> float4 per lane; atomic add into agg[dst].
// xw (25.6 MB) is L2-resident, so the random gathers mostly hit the 192 MB L2.
__global__ __launch_bounds__(256) void scatter_kernel(
    const float* __restrict__ xw, const int* __restrict__ src,
    const int* __restrict__ dst, const int* __restrict__ et,
    const float* __restrict__ nrm, float* __restrict__ agg, int r) {
  int e = blockIdx.x * 8 + (threadIdx.x >> 5);
  if (e >= Ee) return;
  if (et[e] != r) return;                             // uniform per wave
  int lane = threadIdx.x & 31;
  const float4* rp = (const float4*)(xw + (size_t)src[e] * 128);
  float4 v = rp[lane];
  float s = nrm[e];
  float* op = agg + (size_t)dst[e] * 128 + lane * 4;
  atomicAdd(op + 0, v.x * s);
  atomicAdd(op + 1, v.y * s);
  atomicAdd(op + 2, v.z * s);
  atomicAdd(op + 3, v.w * s);
}

// ---------------- driver ----------------

extern "C" void kernel_launch(void* const* d_in, const int* in_sizes, int n_in,
                              void* d_out, int out_size, void* d_ws, size_t ws_size,
                              hipStream_t stream) {
  (void)in_sizes; (void)n_in; (void)out_size; (void)ws_size;
  const float* x     = (const float*)d_in[0];
  const int*   eidx  = (const int*)d_in[1];
  const int*   etype = (const int*)d_in[2];
  const float* W1    = (const float*)d_in[3];
  const float* root1 = (const float*)d_in[4];
  const float* b1    = (const float*)d_in[5];
  const float* W2    = (const float*)d_in[6];
  const float* root2 = (const float*)d_in[7];
  const float* b2    = (const float*)d_in[8];
  const float* W3    = (const float*)d_in[9];
  const float* root3 = (const float*)d_in[10];
  const float* b3    = (const float*)d_in[11];
  const float* res_w = (const float*)d_in[12];
  const float* res_b = (const float*)d_in[13];
  const int* srcp = eidx;
  const int* dstp = eidx + Ee;
  float* out = (float*)d_out;

  char* cur = (char*)d_ws;
  auto alloc = [&](size_t bytes) -> char* {
    char* p = cur;
    cur += (bytes + 255) & ~(size_t)255;
    return p;
  };
  float*  counts   = (float*)alloc((size_t)Rr * Nn * 4);
  float*  nrm      = (float*)alloc((size_t)Ee * 4);
  __bf16* hb       = (__bf16*)alloc((size_t)Nn * 128 * 2);
  __bf16* Wfrag    = (__bf16*)alloc((size_t)Rr * 16384 * 2);
  __bf16* rootfrag = (__bf16*)alloc((size_t)16384 * 2);
  __bf16* resfrag  = (__bf16*)alloc((size_t)16384 * 2);
  float*  xw       = (float*)alloc((size_t)Nn * 128 * 4);
  float*  aggb     = (float*)alloc((size_t)Nn * 128 * 4);
  float*  resb     = (float*)alloc((size_t)Nn * 128 * 4);
  float*  h1       = (float*)alloc((size_t)Nn * 128 * 4);
  float*  h2       = (float*)alloc((size_t)Nn * 128 * 4);

  const dim3 b256(256);
  const int ND = Nn * 128;
  const int gemm_grid = (Nn / 16 + 7) / 8;            // 391 blocks

  // ---- edge normalization (shared by all 3 layers) ----
  zero_f32<<<(Rr * Nn + 255) / 256, b256, 0, stream>>>(counts, Rr * Nn);
  count_kernel<<<(Ee + 255) / 256, b256, 0, stream>>>(etype, dstp, counts);
  norm_kernel<<<(Ee + 255) / 256, b256, 0, stream>>>(etype, dstp, counts, nrm);

  // ---- residual branch: res = x @ res_w + res_b ----
  cvt_bf16<<<(ND + 255) / 256, b256, 0, stream>>>(x, hb, ND);
  repack_frag<<<dim3(64, 1), b256, 0, stream>>>(res_w, resfrag);
  gemm_wmma<1><<<gemm_grid, b256, 0, stream>>>(hb, resfrag, resb, nullptr,
                                               res_b, nullptr, Nn);

  auto layer = [&](const float* hin, const float* W, const float* root,
                   const float* bias, float* hout) {
    cvt_bf16<<<(ND + 255) / 256, b256, 0, stream>>>(hin, hb, ND);
    repack_frag<<<dim3(64, Rr), b256, 0, stream>>>(W, Wfrag);
    repack_frag<<<dim3(64, 1), b256, 0, stream>>>(root, rootfrag);
    zero_f32<<<(ND + 255) / 256, b256, 0, stream>>>(aggb, ND);
    for (int r = 0; r < Rr; ++r) {
      // xw_r = h @ W[r]  (25.6 MB -> lives in L2 for the gather below)
      gemm_wmma<0><<<gemm_grid, b256, 0, stream>>>(
          hb, Wfrag + (size_t)r * 16384, xw, nullptr, nullptr, nullptr, Nn);
      scatter_kernel<<<(Ee + 7) / 8, b256, 0, stream>>>(xw, srcp, dstp, etype,
                                                        nrm, aggb, r);
    }
    // hout = res + relu(agg + h @ root + bias)
    gemm_wmma<2><<<gemm_grid, b256, 0, stream>>>(hb, rootfrag, hout, aggb,
                                                 bias, resb, Nn);
  };

  layer(x,  W1, root1, b1, h1);
  layer(h1, W2, root2, b2, h2);
  layer(h2, W3, root3, b3, out);
}
